// BERTEmSelfAttention_54906861912075
// MI455X (gfx1250) — matrix operand order repaired and checked by reading, not compiled, not run on hardware
//
#include <hip/hip_runtime.h>
#include <hip/hip_bf16.h>

typedef __bf16 bf16;
typedef __attribute__((ext_vector_type(16))) __bf16 v16bf;
typedef __attribute__((ext_vector_type(8)))  __bf16 v8bf;
typedef __attribute__((ext_vector_type(8)))  float  v8f;

#define WMMA_BF16(A, B, C) \
    __builtin_amdgcn_wmma_f32_16x16x32_bf16(false, (A), false, (B), (short)0, (C), false, false)

// ---------------------------------------------------------------------------
// Problem constants
// ---------------------------------------------------------------------------
static constexpr int Bc = 4;
static constexpr int Sc = 2048;
static constexpr int Hc = 1024;
static constexpr int NHc = 16;
static constexpr int HDc = 64;
static constexpr int Mrows = Bc * Sc;  // 8192

// Fragment-workspace strides (bf16 elements)
//   Qf: [b*16+head][qsub(128)][ks(2)][512]   (A-layout frags for QK^T)
//   Kf: [b*16+head][ksub(128)][ks(2)][512]   (B-layout frags for QK^T)
//   Vf: [b*16+head][ktile32(64)][j(4)][512]  (B-layout frags for PV)
static constexpr size_t QK_BH_STRIDE = 128 * 1024;
static constexpr size_t V_BH_STRIDE  = 64 * 2048;

// ---------------------------------------------------------------------------
// Kernel 1: fuse b + emb[index] into per-channel bias vectors (3 x 1024 f32)
// ---------------------------------------------------------------------------
__global__ void fuse_bias_kernel(const float* __restrict__ bq,
                                 const float* __restrict__ bk,
                                 const float* __restrict__ bv,
                                 const float* __restrict__ qe,
                                 const float* __restrict__ ke,
                                 const float* __restrict__ ve,
                                 const int* __restrict__ idx,
                                 float* __restrict__ bias3) {
    int i = blockIdx.x * blockDim.x + threadIdx.x;
    int t = idx[0];
    if (i < Hc) {
        bias3[i]          = bq[i] + qe[(size_t)t * Hc + i];
        bias3[Hc + i]     = bk[i] + ke[(size_t)t * Hc + i];
        bias3[2 * Hc + i] = bv[i] + ve[(size_t)t * Hc + i];
    }
}

// ---------------------------------------------------------------------------
// Kernel 2: QKV projection GEMM with bf16 WMMA.
// C[8192,1024] = hidden[8192,1024] * W[1024,1024] + bias.
// Epilogue stores bf16 in WMMA fragment order; per 16x16 tile the 8 stores
// are a single strided run (Q/K) or one packed v8bf store (V).
// ---------------------------------------------------------------------------
__global__ __launch_bounds__(256) void qkv_gemm_kernel(
    const float* __restrict__ hidden,
    const float* __restrict__ Wq, const float* __restrict__ Wk,
    const float* __restrict__ Wv,
    const float* __restrict__ bias3,
    bf16* __restrict__ Qf, bf16* __restrict__ Kf, bf16* __restrict__ Vf) {
    __shared__ alignas(32) bf16 ldsA[8 * 512];  // 8 M-subtiles of 16x32, frag order
    __shared__ alignas(32) bf16 ldsB[4 * 512];  // 4 N-subtiles of 32x16, frag order

    const int tid  = threadIdx.x;
    const int lane = tid & 31;
    const int wv   = tid >> 5;
    const int wr   = wv >> 1;   // 0..3 (M)
    const int wc   = wv & 1;    // 0..1 (N)
    const int M0   = blockIdx.x * 128;
    const int N0   = blockIdx.y * 64;
    const int mat  = blockIdx.z;

    const float* W    = (mat == 0) ? Wq : (mat == 1) ? Wk : Wv;
    const float* bias = bias3 + mat * Hc;

    v8f acc[2][2];
#pragma unroll
    for (int ti = 0; ti < 2; ++ti)
#pragma unroll
        for (int tj = 0; tj < 2; ++tj) acc[ti][tj] = (v8f){};

    for (int kt = 0; kt < Hc; kt += 32) {
        __syncthreads();
        // Stage A tile 128x32 f32 -> bf16 in WMMA A-fragment order.
        for (int idx = tid; idx < 4096; idx += 256) {
            int st = idx >> 9, r = idx & 511, ln = r >> 4, i = r & 15;
            int m  = st * 16 + (ln & 15);
            int g  = ln >> 4, p = i >> 1, hh = i & 1;
            int kl = ((p < 4) ? 2 * p : 2 * p + 8) + 8 * g + hh;
            ldsA[idx] = (bf16)hidden[(size_t)(M0 + m) * Hc + kt + kl];
        }
        // Stage B tile 32x64 -> bf16 in B-fragment order.
        for (int idx = tid; idx < 2048; idx += 256) {
            int st = idx >> 9, r = idx & 511, ln = r >> 4, i = r & 15;
            int n  = st * 16 + (ln & 15);
            int kl = (i & 7) + 8 * (ln >> 4) + 16 * (i >> 3);
            ldsB[idx] = (bf16)W[(size_t)(kt + kl) * Hc + N0 + n];
        }
        __syncthreads();

        v16bf a0 = *(const v16bf*)&ldsA[(2 * wr) * 512 + lane * 16];
        v16bf a1 = *(const v16bf*)&ldsA[(2 * wr + 1) * 512 + lane * 16];
        v16bf b0 = *(const v16bf*)&ldsB[(2 * wc) * 512 + lane * 16];
        v16bf b1 = *(const v16bf*)&ldsB[(2 * wc + 1) * 512 + lane * 16];

        acc[0][0] = WMMA_BF16(a0, b0, acc[0][0]);
        acc[0][1] = WMMA_BF16(a0, b1, acc[0][1]);
        acc[1][0] = WMMA_BF16(a1, b0, acc[1][0]);
        acc[1][1] = WMMA_BF16(a1, b1, acc[1][1]);
    }

    // Epilogue. C/D layout: row m = v + 8*gc, col n = nl (within tile).
    const int gc    = lane >> 4;
    const int nl    = lane & 15;
    const int b_idx = M0 >> 11;  // batch (tiles never straddle batches)

#pragma unroll
    for (int ti = 0; ti < 2; ++ti) {
        const int msub   = 2 * wr + ti;
        const int s_base = (M0 & (Sc - 1)) + msub * 16 + 8 * gc;  // token, v=0
#pragma unroll
        for (int tj = 0; tj < 2; ++tj) {
            const int nsub = 2 * wc + tj;
            const int n    = N0 + nsub * 16 + nl;
            const int head = n >> 6;
            const int c    = n & 63;
            const int bh   = b_idx * NHc + head;
            const float bb = bias[n];
            const v8f&  a  = acc[ti][tj];
            if (mat == 2) {
                // V: 8 values are contiguous bf16 -> one 16B packed store.
                const int kt32 = s_base >> 5;
                const int kkb  = s_base & 31;           // in {0,8,16,24}
                const int gB   = (kkb >> 3) & 1, ib = (kkb >> 4) * 8;
                const int laneB = (c & 15) | (gB << 4);
                v8bf pv;
#pragma unroll
                for (int v = 0; v < 8; ++v) pv[v] = (bf16)(a[v] + bb);
                *(v8bf*)&Vf[((size_t)bh * 64 + kt32) * 2048 +
                            (c >> 4) * 512 + laneB * 16 + ib] = pv;
            } else {
                // Q (A-layout) / K (B-layout): stores at addr0 + v*16 elems.
                const int ks = c >> 5, kl = c & 31;
                int off16, ii;
                if (mat == 0) {
                    const int gA = (kl >> 3) & 1;
                    const int kp = kl - 8 * gA;
                    off16 = gA << 4;
                    ii    = (kp < 8) ? kp : (kp - 8);
                } else {
                    off16 = ((kl >> 3) & 1) << 4;
                    ii    = (kl & 7) + 8 * (kl >> 4);
                }
                bf16* dst = ((mat == 0) ? Qf : Kf) +
                            ((size_t)bh * 128 + (s_base >> 4)) * 1024 +
                            ks * 512 + (size_t)(8 * gc + off16) * 16 + ii;
#pragma unroll
                for (int v = 0; v < 8; ++v) dst[v * 16] = (bf16)(a[v] + bb);
            }
        }
    }
}

// ---------------------------------------------------------------------------
// Kernel 3: fused flash attention, fragment loads straight from global.
// One block per (b, head, 128-query tile); 8 independent waves, each owning
// 16 query rows. No __syncthreads; only LDS use is the per-wave P transpose.
// ---------------------------------------------------------------------------
__global__ __launch_bounds__(256) void flash_attn_kernel(
    const bf16* __restrict__ Qf, const bf16* __restrict__ Kf,
    const bf16* __restrict__ Vf, const float* __restrict__ mask,
    float* __restrict__ out) {
    __shared__ alignas(32) bf16 ldsP[8 * 512];  // per-wave P A-frag (8KB)

    const int tid  = threadIdx.x;
    const int lane = tid & 31;
    const int wv   = tid >> 5;
    const int qt   = blockIdx.x & 15;
    const int head = (blockIdx.x >> 4) & 15;
    const int b    = blockIdx.x >> 8;
    const int bh   = b * NHc + head;

    // Per-wave Q fragments: 32 contiguous bytes per lane -> direct v16bf load.
    const int qsub = qt * 8 + wv;
    const bf16* qbase = Qf + ((size_t)bh * 128 + qsub) * 1024 + lane * 16;
    v16bf qa0 = *(const v16bf*)qbase;
    v16bf qa1 = *(const v16bf*)(qbase + 512);

    v8f acc0 = {}, acc1 = {}, acc2 = {}, acc3 = {};
    float rm[8], rl[8];
#pragma unroll
    for (int v = 0; v < 8; ++v) { rm[v] = -1e30f; rl[v] = 0.0f; }

    const int   g     = lane >> 4;
    const int   nl    = lane & 15;
    const float scale = 0.125f;  // 1/sqrt(64)

    const bf16*  kbh  = Kf + (size_t)bh * QK_BH_STRIDE + lane * 16;
    const bf16*  vbh  = Vf + (size_t)bh * V_BH_STRIDE + lane * 16;
    const float* mrow = mask + (size_t)b * Sc + nl;

    for (int kb = 0; kb < Sc / 32; ++kb) {
        const bf16* kbase = kbh + (size_t)kb * 2048;
        const bf16* vbase = vbh + (size_t)kb * 2048;

        // K^T fragments (nsub, ks) at consecutive 512-element offsets.
        v16bf kf00 = *(const v16bf*)(kbase);
        v16bf kf01 = *(const v16bf*)(kbase + 512);
        v16bf kf10 = *(const v16bf*)(kbase + 1024);
        v16bf kf11 = *(const v16bf*)(kbase + 1536);
        // V fragments (d-subtile j).
        v16bf vf0 = *(const v16bf*)(vbase);
        v16bf vf1 = *(const v16bf*)(vbase + 512);
        v16bf vf2 = *(const v16bf*)(vbase + 1024);
        v16bf vf3 = *(const v16bf*)(vbase + 1536);

        // Branchless prefetch of next key tile (wraps in-bounds on last iter).
        const size_t pnext = (size_t)((kb + 1) & 63) * 2048;
        __builtin_prefetch(kbh + pnext, 0, 1);
        __builtin_prefetch(vbh + pnext, 0, 1);

        v8f d0 = {}, d1 = {};
        d0 = WMMA_BF16(qa0, kf00, d0);
        d0 = WMMA_BF16(qa1, kf01, d0);
        d1 = WMMA_BF16(qa0, kf10, d1);
        d1 = WMMA_BF16(qa1, kf11, d1);

        const float mk0 = mrow[kb * 32];
        const float mk1 = mrow[kb * 32 + 16];

        float p0[8], p1[8];
#pragma unroll
        for (int v = 0; v < 8; ++v) {
            float s0 = d0[v] * scale + mk0;
            float s1 = d1[v] * scale + mk1;
            float x = fmaxf(s0, s1);
            x = fmaxf(x, __shfl_xor(x, 1));
            x = fmaxf(x, __shfl_xor(x, 2));
            x = fmaxf(x, __shfl_xor(x, 4));
            x = fmaxf(x, __shfl_xor(x, 8));
            float mnew = fmaxf(rm[v], x);
            float corr = __expf(rm[v] - mnew);
            p0[v] = __expf(s0 - mnew);
            p1[v] = __expf(s1 - mnew);
            float sum = p0[v] + p1[v];
            sum += __shfl_xor(sum, 1);
            sum += __shfl_xor(sum, 2);
            sum += __shfl_xor(sum, 4);
            sum += __shfl_xor(sum, 8);
            rl[v] = rl[v] * corr + sum;
            rm[v] = mnew;
            acc0[v] *= corr; acc1[v] *= corr; acc2[v] *= corr; acc3[v] *= corr;
        }

        // Transpose P (C/D layout, f32) -> A-fragment layout (bf16) via LDS.
        auto stageP = [&](float val, int m, int kk) {
            int gA = (kk >> 3) & 1;
            int kp = kk - 8 * gA;
            int iA = (kp < 8) ? kp : (kp - 8);
            int laneA = (m & 15) | (gA << 4);
            ldsP[wv * 512 + laneA * 16 + iA] = (bf16)val;
        };
#pragma unroll
        for (int v = 0; v < 8; ++v) {
            int m = v + 8 * g;
            stageP(p0[v], m, nl);
            stageP(p1[v], m, 16 + nl);
        }
        v16bf pa = *(const v16bf*)&ldsP[wv * 512 + lane * 16];

        acc0 = WMMA_BF16(pa, vf0, acc0);
        acc1 = WMMA_BF16(pa, vf1, acc1);
        acc2 = WMMA_BF16(pa, vf2, acc2);
        acc3 = WMMA_BF16(pa, vf3, acc3);
    }

    // Final normalization + store to [B, S, NH*HD] layout.
    float* outp = out + ((size_t)b * Sc + qt * 128 + wv * 16 + 8 * g) * Hc +
                  head * HDc + nl;
#pragma unroll
    for (int v = 0; v < 8; ++v) {
        float inv = 1.0f / rl[v];
        outp[(size_t)v * Hc + 0]  = acc0[v] * inv;
        outp[(size_t)v * Hc + 16] = acc1[v] * inv;
        outp[(size_t)v * Hc + 32] = acc2[v] * inv;
        outp[(size_t)v * Hc + 48] = acc3[v] * inv;
    }
}

// ---------------------------------------------------------------------------
// Launcher
// ---------------------------------------------------------------------------
extern "C" void kernel_launch(void* const* d_in, const int* in_sizes, int n_in,
                              void* d_out, int out_size, void* d_ws, size_t ws_size,
                              hipStream_t stream) {
    const float* hidden = (const float*)d_in[0];
    const float* mask   = (const float*)d_in[1];
    const float* Wq     = (const float*)d_in[2];
    const float* bq     = (const float*)d_in[3];
    const float* Wk     = (const float*)d_in[4];
    const float* bk     = (const float*)d_in[5];
    const float* Wv     = (const float*)d_in[6];
    const float* bv     = (const float*)d_in[7];
    const float* qe     = (const float*)d_in[8];
    const float* ke     = (const float*)d_in[9];
    const float* ve     = (const float*)d_in[10];
    const int*   idx    = (const int*)d_in[11];
    float* out = (float*)d_out;

    char*  ws    = (char*)d_ws;
    float* bias3 = (float*)ws;                  // 3*1024 f32
    bf16*  Qf    = (bf16*)(ws + 65536);         // 8M bf16 (16MB)
    bf16*  Kf    = Qf + (size_t)Mrows * Hc;     // 8M bf16
    bf16*  Vf    = Kf + (size_t)Mrows * Hc;     // 8M bf16

    fuse_bias_kernel<<<4, 256, 0, stream>>>(bq, bk, bv, qe, ke, ve, idx, bias3);

    dim3 gemm_grid(Mrows / 128, Hc / 64, 3);    // 64 x 16 x 3
    qkv_gemm_kernel<<<gemm_grid, 256, 0, stream>>>(hidden, Wq, Wk, Wv, bias3,
                                                   Qf, Kf, Vf);

    int attn_blocks = Bc * NHc * (Sc / 128);    // 1024
    flash_attn_kernel<<<attn_blocks, 256, 0, stream>>>(Qf, Kf, Vf, mask, out);
}
